// Decoder_7387343749191
// MI455X (gfx1250) — matrix operand order repaired and checked by reading, not compiled, not run on hardware
//
#include <hip/hip_runtime.h>
#include <cstdint>
#include <cstddef>

// ---------------------------------------------------------------------------
// Problem constants (from the reference)
// ---------------------------------------------------------------------------
#define NB     32      // batch
#define NTENC  256     // encoder time steps
#define NE     512     // encoder embedding
#define NH     512     // GRU hidden
#define NA     256     // attention dim
#define NMELR  400     // N_MELS * R
#define NSTEP  200     // decoder scan steps
#define KPRE   416     // 400 padded to multiple of 32

typedef float v8f   __attribute__((ext_vector_type(8)));
typedef __bf16 v16bf __attribute__((ext_vector_type(16)));

union V16U { uint4 u[2]; v16bf v; };

// ---------------------------------------------------------------------------
// WMMA bf16 GEMM with B-fragment reuse:  C[M,N] = act(A[M,K] @ B[K,N] + bias)
//   Each wave owns a 32x16 output slab (two 16x16 accumulators) so every
//   packed weight fragment is loaded ONCE and fed to TWO wmma ops.
//   A : bf16 row-major, row stride lda (elements), lda % 8 == 0
//   Bp: pre-packed bf16 tiles in WMMA B-fragment register order:
//       block (kt*ntiles + nt) of 512 elems; lane l holds 16 contiguous elems
//       = W[kt*32 + (l>=16?16:0) + i , nt*16 + (l&15)]
//   grid = (N/16, M/32), block = 32 (one wave)
//   Optional f32 output (stride ldc) and/or bf16 output (stride ldcb).
// ---------------------------------------------------------------------------
__global__ void gemm_wmma_bf16(const __bf16* __restrict__ A, int lda,
                               const __bf16* __restrict__ Bp,
                               const float* __restrict__ bias,
                               int K, int N, int relu,
                               float* __restrict__ Cf, int ldc,
                               __bf16* __restrict__ Cb, int ldcb)
{
    const int lane   = threadIdx.x;        // 0..31
    const int nt     = blockIdx.x;
    const int mt2    = blockIdx.y;         // 32-row slab index
    const int ntiles = N >> 4;
    const int hi     = lane >> 4;          // 0 or 1 (K half select)
    const int m0     = mt2 * 32 + (lane & 15);
    const int n      = nt * 16 + (lane & 15);

    const __bf16* arow0 = A + (size_t)m0 * lda + hi * 16;
    const __bf16* arow1 = arow0 + (size_t)16 * lda;
    const __bf16* bptr  = Bp + (size_t)nt * 512 + (size_t)lane * 16;
    const size_t  bstep = (size_t)ntiles * 512;

    v8f acc0 = {};
    v8f acc1 = {};
    const int nkt = K >> 5;
    for (int kt = 0; kt < nkt; ++kt) {
        V16U a0, a1, b;
        const __bf16* ap0 = arow0 + kt * 32;
        const __bf16* ap1 = arow1 + kt * 32;
        b.u[0]  = *(const uint4*)(bptr);
        b.u[1]  = *(const uint4*)(bptr + 8);
        a0.u[0] = *(const uint4*)(ap0);
        a0.u[1] = *(const uint4*)(ap0 + 8);
        a1.u[0] = *(const uint4*)(ap1);
        a1.u[1] = *(const uint4*)(ap1 + 8);
        bptr += bstep;
        __builtin_prefetch(bptr, 0, 1);    // next weight tile (global_prefetch_b8)
        acc0 = __builtin_amdgcn_wmma_f32_16x16x32_bf16(
                   false, a0.v, false, b.v, (short)0, acc0, false, false);
        acc1 = __builtin_amdgcn_wmma_f32_16x16x32_bf16(
                   false, a1.v, false, b.v, (short)0, acc1, false, false);
    }

    const float bv = bias ? bias[n] : 0.0f;
#pragma unroll
    for (int v = 0; v < 8; ++v) {
        const int row0 = mt2 * 32 + hi * 8 + v;   // C layout: lane>=16 -> M+8
        const int row1 = row0 + 16;
        float v0 = acc0[v] + bv;
        float v1 = acc1[v] + bv;
        if (relu) { v0 = fmaxf(v0, 0.0f); v1 = fmaxf(v1, 0.0f); }
        if (Cf) {
            Cf[(size_t)row0 * ldc + n] = v0;
            Cf[(size_t)row1 * ldc + n] = v1;
        }
        if (Cb) {
            Cb[(size_t)row0 * ldcb + n] = (__bf16)v0;
            Cb[(size_t)row1 * ldcb + n] = (__bf16)v1;
        }
    }
}

// ---------------------------------------------------------------------------
// Weight packer: W[Korig,N] f32 row-major -> packed bf16 WMMA B tiles,
// K padded to Kpad (zeros).  One thread per packed element (Kpad*N total).
// ---------------------------------------------------------------------------
__global__ void pack_weight(const float* __restrict__ W, int Korig, int Kpad,
                            int N, __bf16* __restrict__ out)
{
    const size_t idx   = (size_t)blockIdx.x * blockDim.x + threadIdx.x;
    const size_t total = (size_t)Kpad * N;
    if (idx >= total) return;
    const int ntiles = N >> 4;
    const size_t blk = idx >> 9;          // 512-elem tile
    const int within = (int)(idx & 511);
    const int lane   = within >> 4;
    const int elem   = within & 15;
    const int kt = (int)(blk / ntiles);
    const int nt = (int)(blk % ntiles);
    const int k  = kt * 32 + ((lane >> 4) << 4) + elem;
    const int n  = nt * 16 + (lane & 15);
    const float v = (k < Korig) ? W[(size_t)k * N + n] : 0.0f;
    out[idx] = (__bf16)v;
}

// f32 -> bf16 elementwise
__global__ void cvt_f32_bf16(const float* __restrict__ in, __bf16* __restrict__ out,
                             size_t n)
{
    const size_t i = (size_t)blockIdx.x * blockDim.x + threadIdx.x;
    if (i < n) out[i] = (__bf16)in[i];
}

// Teacher-forced decoder inputs -> bf16, shifted right one group, K padded.
// Layout: xin[NSTEP][NB][KPRE]
__global__ void pack_xin(const float* __restrict__ dec, __bf16* __restrict__ xin)
{
    const size_t i = (size_t)blockIdx.x * blockDim.x + threadIdx.x;
    const size_t total = (size_t)NSTEP * NB * KPRE;
    if (i >= total) return;
    const int t = (int)(i / (NB * KPRE));
    const int r = (int)(i % (NB * KPRE));
    const int b = r / KPRE;
    const int k = r % KPRE;
    float v = 0.0f;
    if (t > 0 && k < NMELR)
        v = dec[(size_t)b * (NSTEP * NMELR) + (size_t)(t - 1) * NMELR + k];
    xin[i] = (__bf16)v;
}

__global__ void zero_f32(float* p, size_t n)
{
    const size_t i = (size_t)blockIdx.x * blockDim.x + threadIdx.x;
    if (i < n) p[i] = 0.0f;
}
__global__ void zero_bf16(__bf16* p, size_t n)
{
    const size_t i = (size_t)blockIdx.x * blockDim.x + threadIdx.x;
    if (i < n) p[i] = (__bf16)0.0f;
}

// ---------------------------------------------------------------------------
// GRU combine: h = z*hprev + (1-z)*tanh(xh + r*hh),  gates from gx/gh [NB,3H].
// Writes f32 carry + bf16 copy; optional second bf16 copy (into a concat
// buffer, stride ld2); optional residual res = resbase + h (f32 and/or bf16).
// 32*512 threads.
// ---------------------------------------------------------------------------
__global__ void gru_combine(const float* __restrict__ gx, const float* __restrict__ gh,
                            const float* __restrict__ hprev,
                            float* __restrict__ h_f, __bf16* __restrict__ h_b,
                            __bf16* __restrict__ h_b2, int ld2,
                            const float* __restrict__ resbase,
                            float* __restrict__ res_f, __bf16* __restrict__ res_b)
{
    const int i = blockIdx.x * blockDim.x + threadIdx.x;
    if (i >= NB * NH) return;
    const int b = i >> 9;              // NH == 512
    const int j = i & (NH - 1);
    const size_t g = (size_t)b * (3 * NH);
    const float xz = gx[g + j], xr = gx[g + NH + j], xh = gx[g + 2 * NH + j];
    const float hz = gh[g + j], hr = gh[g + NH + j], hh = gh[g + 2 * NH + j];
    const float z  = 1.0f / (1.0f + expf(-(xz + hz)));
    const float r  = 1.0f / (1.0f + expf(-(xr + hr)));
    const float hc = tanhf(xh + r * hh);
    const float h  = z * hprev[i] + (1.0f - z) * hc;
    h_f[i] = h;
    h_b[i] = (__bf16)h;
    if (h_b2) h_b2[(size_t)b * ld2 + j] = (__bf16)h;
    if (resbase) {
        const float rs = resbase[i] + h;
        if (res_f) res_f[i] = rs;
        if (res_b) res_b[i] = (__bf16)rs;
    }
}

// score[b,t] = sum_a tanh(keys[b,t,a] + q[b,a]) * Wv[a] + bv   (32x256 threads)
__global__ void att_score(const float* __restrict__ keys, const float* __restrict__ q,
                          const float* __restrict__ Wv, const float* __restrict__ bv,
                          float* __restrict__ score)
{
    const int i = blockIdx.x * blockDim.x + threadIdx.x;
    if (i >= NB * NTENC) return;
    const int b = i >> 8;
    const float* kp = keys + (size_t)i * NA;
    const float* qp = q + (size_t)b * NA;
    float s = bv[0];
    for (int a = 0; a < NA; ++a)
        s += tanhf(kp[a] + qp[a]) * Wv[a];
    score[i] = s;
}

// Softmax over T_ENC per batch; one block per batch, LDS tree reductions.
// Writes align (ws) and the alignments output slice in d_out.
__global__ void att_softmax(const float* __restrict__ score, float* __restrict__ align,
                            float* __restrict__ aligns_out, int step)
{
    __shared__ float red[NTENC];
    const int b = blockIdx.x;
    const int t = threadIdx.x;
    const float v = score[b * NTENC + t];
    red[t] = v;
    __syncthreads();
    for (int s = NTENC / 2; s > 0; s >>= 1) {
        if (t < s) red[t] = fmaxf(red[t], red[t + s]);
        __syncthreads();
    }
    const float mx = red[0];
    __syncthreads();
    const float e = expf(v - mx);
    red[t] = e;
    __syncthreads();
    for (int s = NTENC / 2; s > 0; s >>= 1) {
        if (t < s) red[t] += red[t + s];
        __syncthreads();
    }
    const float a = e / red[0];
    align[b * NTENC + t] = a;
    aligns_out[(size_t)b * (NSTEP * NTENC) + (size_t)step * NTENC + t] = a;
}

// attn[b,e] = sum_t align[b,t] * memory[b,t,e]; bf16 copies straight into the
// two concat buffers (ax cols 128.., pc cols 0..). 32*512 threads.
__global__ void att_context(const float* __restrict__ align, const float* __restrict__ memory,
                            __bf16* __restrict__ axb, __bf16* __restrict__ pcb)
{
    const int i = blockIdx.x * blockDim.x + threadIdx.x;
    if (i >= NB * NE) return;
    const int b = i >> 9;                 // NE == 512
    const int e = i & (NE - 1);
    const float* al = align + (size_t)b * NTENC;
    const float* mp = memory + (size_t)b * NTENC * NE + e;
    float s = 0.0f;
    for (int t = 0; t < NTENC; ++t)
        s += al[t] * mp[(size_t)t * NE];
    axb[(size_t)b * 640 + 128 + e]  = (__bf16)s;
    pcb[(size_t)b * 1024 + e]       = (__bf16)s;
}

// ---------------------------------------------------------------------------
// Host orchestration
// ---------------------------------------------------------------------------
static inline void run_gemm(hipStream_t s, const __bf16* A, int lda, const __bf16* Bp,
                            const float* bias, int K, int N, int M, int relu,
                            float* Cf, int ldc, __bf16* Cb, int ldcb)
{
    dim3 grid(N / 16, M / 32);
    gemm_wmma_bf16<<<grid, dim3(32), 0, s>>>(A, lda, Bp, bias, K, N, relu, Cf, ldc, Cb, ldcb);
}

extern "C" void kernel_launch(void* const* d_in, const int* in_sizes, int n_in,
                              void* d_out, int out_size, void* d_ws, size_t ws_size,
                              hipStream_t stream)
{
    (void)in_sizes; (void)n_in; (void)out_size; (void)ws_size;

    const float* memory = (const float*)d_in[0];
    const float* dec    = (const float*)d_in[1];
    const float* W1  = (const float*)d_in[2];
    const float* b1  = (const float*)d_in[3];
    const float* W2  = (const float*)d_in[4];
    const float* b2  = (const float*)d_in[5];
    const float* Wm  = (const float*)d_in[6];
    const float* Wq  = (const float*)d_in[7];
    const float* Wv  = (const float*)d_in[8];
    const float* bv  = (const float*)d_in[9];
    const float* aWk = (const float*)d_in[10];
    const float* aWr = (const float*)d_in[11];
    const float* ab  = (const float*)d_in[12];
    const float* Wp  = (const float*)d_in[13];
    const float* bp  = (const float*)d_in[14];
    const float* d1Wk = (const float*)d_in[15];
    const float* d1Wr = (const float*)d_in[16];
    const float* d1b  = (const float*)d_in[17];
    const float* d2Wk = (const float*)d_in[18];
    const float* d2Wr = (const float*)d_in[19];
    const float* d2b  = (const float*)d_in[20];
    const float* Wo  = (const float*)d_in[21];
    const float* bo  = (const float*)d_in[22];

    float* dec_out    = (float*)d_out;                               // [32,1000,80]
    float* aligns_out = (float*)d_out + (size_t)NB * 1000 * 80;      // [32,200,256]

    // ---- workspace carve (256B aligned) ----
    char* base = (char*)d_ws;
    size_t cur = 0;
    auto carve = [&](size_t bytes) -> char* {
        char* p = base + cur;
        cur = (cur + bytes + 255) & ~(size_t)255;
        return p;
    };
    __bf16* xin   = (__bf16*)carve((size_t)NSTEP * NB * KPRE * 2);
    __bf16* memb  = (__bf16*)carve((size_t)NB * NTENC * NE * 2);
    float*  keys  = (float*) carve((size_t)NB * NTENC * NA * 4);
    __bf16* W1p   = (__bf16*)carve((size_t)KPRE * 256 * 2);
    __bf16* W2p   = (__bf16*)carve((size_t)256 * 128 * 2);
    __bf16* Wmp   = (__bf16*)carve((size_t)NE * NA * 2);
    __bf16* Wqp   = (__bf16*)carve((size_t)NH * NA * 2);
    __bf16* aWkp  = (__bf16*)carve((size_t)640 * 1536 * 2);
    __bf16* aWrp  = (__bf16*)carve((size_t)NH * 1536 * 2);
    __bf16* Wpp   = (__bf16*)carve((size_t)1024 * NH * 2);
    __bf16* d1Wkp = (__bf16*)carve((size_t)NH * 1536 * 2);
    __bf16* d1Wrp = (__bf16*)carve((size_t)NH * 1536 * 2);
    __bf16* d2Wkp = (__bf16*)carve((size_t)NH * 1536 * 2);
    __bf16* d2Wrp = (__bf16*)carve((size_t)NH * 1536 * 2);
    __bf16* Wop   = (__bf16*)carve((size_t)NH * NMELR * 2);
    __bf16* p1    = (__bf16*)carve((size_t)NB * 256 * 2);
    __bf16* axb   = (__bf16*)carve((size_t)NB * 640 * 2);   // [p2 | attn]
    __bf16* pcb   = (__bf16*)carve((size_t)NB * 1024 * 2);  // [attn | h_a]
    float*  gx    = (float*) carve((size_t)NB * 1536 * 4);
    float*  gh    = (float*) carve((size_t)NB * 1536 * 4);
    float*  haf   = (float*) carve((size_t)NB * NH * 4);
    __bf16* hab   = (__bf16*)carve((size_t)NB * NH * 2);
    float*  h1f   = (float*) carve((size_t)NB * NH * 4);
    __bf16* h1b   = (__bf16*)carve((size_t)NB * NH * 2);
    float*  h2f   = (float*) carve((size_t)NB * NH * 4);
    __bf16* h2b   = (__bf16*)carve((size_t)NB * NH * 2);
    float*  qbuf  = (float*) carve((size_t)NB * NA * 4);
    float*  score = (float*) carve((size_t)NB * NTENC * 4);
    float*  align = (float*) carve((size_t)NB * NTENC * 4);
    float*  projf = (float*) carve((size_t)NB * NH * 4);
    __bf16* projb = (__bf16*)carve((size_t)NB * NH * 2);
    float*  r1f   = (float*) carve((size_t)NB * NH * 4);
    __bf16* r1b   = (__bf16*)carve((size_t)NB * NH * 2);
    __bf16* r2b   = (__bf16*)carve((size_t)NB * NH * 2);

    const int TPB = 256;
    auto blocks = [&](size_t n) { return (unsigned)((n + TPB - 1) / TPB); };

    // ---- one-time prep ----
    pack_xin<<<blocks((size_t)NSTEP * NB * KPRE), TPB, 0, stream>>>(dec, xin);
    cvt_f32_bf16<<<blocks((size_t)NB * NTENC * NE), TPB, 0, stream>>>(memory, memb,
                                                                      (size_t)NB * NTENC * NE);
    pack_weight<<<blocks((size_t)KPRE * 256), TPB, 0, stream>>>(W1, NMELR, KPRE, 256, W1p);
    pack_weight<<<blocks((size_t)256 * 128), TPB, 0, stream>>>(W2, 256, 256, 128, W2p);
    pack_weight<<<blocks((size_t)NE * NA), TPB, 0, stream>>>(Wm, NE, NE, NA, Wmp);
    pack_weight<<<blocks((size_t)NH * NA), TPB, 0, stream>>>(Wq, NH, NH, NA, Wqp);
    pack_weight<<<blocks((size_t)640 * 1536), TPB, 0, stream>>>(aWk, 640, 640, 1536, aWkp);
    pack_weight<<<blocks((size_t)NH * 1536), TPB, 0, stream>>>(aWr, NH, NH, 1536, aWrp);
    pack_weight<<<blocks((size_t)1024 * NH), TPB, 0, stream>>>(Wp, 1024, 1024, NH, Wpp);
    pack_weight<<<blocks((size_t)NH * 1536), TPB, 0, stream>>>(d1Wk, NH, NH, 1536, d1Wkp);
    pack_weight<<<blocks((size_t)NH * 1536), TPB, 0, stream>>>(d1Wr, NH, NH, 1536, d1Wrp);
    pack_weight<<<blocks((size_t)NH * 1536), TPB, 0, stream>>>(d2Wk, NH, NH, 1536, d2Wkp);
    pack_weight<<<blocks((size_t)NH * 1536), TPB, 0, stream>>>(d2Wr, NH, NH, 1536, d2Wrp);
    pack_weight<<<blocks((size_t)NH * NMELR), TPB, 0, stream>>>(Wo, NH, NH, NMELR, Wop);

    zero_f32<<<blocks((size_t)NB * NH), TPB, 0, stream>>>(haf, (size_t)NB * NH);
    zero_f32<<<blocks((size_t)NB * NH), TPB, 0, stream>>>(h1f, (size_t)NB * NH);
    zero_f32<<<blocks((size_t)NB * NH), TPB, 0, stream>>>(h2f, (size_t)NB * NH);
    zero_bf16<<<blocks((size_t)NB * NH), TPB, 0, stream>>>(hab, (size_t)NB * NH);
    zero_bf16<<<blocks((size_t)NB * NH), TPB, 0, stream>>>(h1b, (size_t)NB * NH);
    zero_bf16<<<blocks((size_t)NB * NH), TPB, 0, stream>>>(h2b, (size_t)NB * NH);
    zero_bf16<<<blocks((size_t)NB * 640), TPB, 0, stream>>>(axb, (size_t)NB * 640);
    zero_bf16<<<blocks((size_t)NB * 1024), TPB, 0, stream>>>(pcb, (size_t)NB * 1024);

    // keys = memory @ Wm   : [8192,512] x [512,256] -> f32 [8192,256]
    run_gemm(stream, memb, NE, Wmp, nullptr, NE, NA, NB * NTENC, 0,
             keys, NA, nullptr, 0);

    // ---- sequential decoder scan ----
    for (int t = 0; t < NSTEP; ++t) {
        const __bf16* xt = xin + (size_t)t * NB * KPRE;

        // prenet
        run_gemm(stream, xt, KPRE, W1p, b1, KPRE, 256, NB, 1, nullptr, 0, p1, 256);
        run_gemm(stream, p1, 256, W2p, b2, 256, 128, NB, 1, nullptr, 0, axb, 640);

        // attention GRU: gx = [p2|attn] @ aWk + ab ; gh = h_a @ aWr
        run_gemm(stream, axb, 640, aWkp, ab, 640, 1536, NB, 0, gx, 1536, nullptr, 0);
        run_gemm(stream, hab, NH, aWrp, nullptr, NH, 1536, NB, 0, gh, 1536, nullptr, 0);
        gru_combine<<<blocks((size_t)NB * NH), TPB, 0, stream>>>(
            gx, gh, haf, haf, hab, pcb + NH, 1024, nullptr, nullptr, nullptr);

        // attention
        run_gemm(stream, hab, NH, Wqp, nullptr, NH, NA, NB, 0, qbuf, NA, nullptr, 0);
        att_score<<<blocks((size_t)NB * NTENC), TPB, 0, stream>>>(keys, qbuf, Wv, bv, score);
        att_softmax<<<NB, NTENC, 0, stream>>>(score, align, aligns_out, t);
        att_context<<<blocks((size_t)NB * NE), TPB, 0, stream>>>(align, memory, axb, pcb);

        // projection
        run_gemm(stream, pcb, 1024, Wpp, bp, 1024, NH, NB, 0, projf, NH, projb, NH);

        // decoder GRU 1 (+residual)
        run_gemm(stream, projb, NH, d1Wkp, d1b, NH, 1536, NB, 0, gx, 1536, nullptr, 0);
        run_gemm(stream, h1b, NH, d1Wrp, nullptr, NH, 1536, NB, 0, gh, 1536, nullptr, 0);
        gru_combine<<<blocks((size_t)NB * NH), TPB, 0, stream>>>(
            gx, gh, h1f, h1f, h1b, nullptr, 0, projf, r1f, r1b);

        // decoder GRU 2 (+residual)
        run_gemm(stream, r1b, NH, d2Wkp, d2b, NH, 1536, NB, 0, gx, 1536, nullptr, 0);
        run_gemm(stream, h2b, NH, d2Wrp, nullptr, NH, 1536, NB, 0, gh, 1536, nullptr, 0);
        gru_combine<<<blocks((size_t)NB * NH), TPB, 0, stream>>>(
            gx, gh, h2f, h2f, h2b, nullptr, 0, r1f, nullptr, r2b);

        // output projection straight into dec_out[b, t*5:(t+1)*5, :]
        run_gemm(stream, r2b, NH, Wop, bo, NH, NMELR, NB, 0,
                 dec_out + (size_t)t * NMELR, NSTEP * NMELR, nullptr, 0);
    }
}
